// CrossAttentionDecoder1Reg_21345987461402
// MI455X (gfx1250) — matrix operand order repaired and checked by reading, not compiled, not run on hardware
//
#include <hip/hip_runtime.h>
#include <hip/hip_bf16.h>
#include <math.h>

// ---------------------------------------------------------------------------
// CrossAttentionDecoder (B=256, S=512, H=512, NH=8, HD=64, L=2, T=6)
//
// Strategy (MI455X / gfx1250, wave32, WMMA):
//  * All big GEMMs via v_wmma_f32_16x16x32_f16 (f16 in, f32 accumulate),
//    templated so the K-loop has zero runtime branches.
//  * KV-projection GEMM (137 GFLOP) uses 4 N-tiles per wave: one A fragment
//    feeds 4 WMMAs from registers (4x A reuse).
//  * K/V stored f16: attention (memory bound) traffic = 268 MB/step
//    @ 23.3 TB/s ~= 11.5 us instead of 23 us in f32.
//  * __builtin_prefetch of next A chunk -> global_prefetch_b8.
// ---------------------------------------------------------------------------

#define Bsz 256
#define Ssz 512
#define Hsz 512
#define NHh 8
#define HDd 64
#define Tst 6

typedef _Float16 half8  __attribute__((ext_vector_type(8)));
typedef _Float16 half16 __attribute__((ext_vector_type(16)));
typedef float    float8 __attribute__((ext_vector_type(8)));
typedef float    fp32x4 __attribute__((ext_vector_type(4)));

// ---------------------------------------------------------------------------
// WMMA GEMM: C[M,N] (+)= A[M,K] * B[N,K]^T + bias[N]
//   A: f16 (f32 when AF32) row-major, lda = K stride
//   B: f16 row-major (N x K) == the reference's W[out,in] layout
//   Cf: optional f32 out (ldc), Ch: optional f16 out (ldh)
// Block = 8 waves, each wave computes NTW consecutive 16x16 N-tiles of one
// M row, reusing its A fragment in registers for NTW WMMAs.
// ---------------------------------------------------------------------------
template <int AF32, int NTW, int ACC>
__global__ void wmma_gemm_kernel(const void* __restrict__ Aptr, int lda,
                                 const _Float16* __restrict__ Bw, int ldb,
                                 const float* __restrict__ bias,
                                 float* __restrict__ Cf, int ldc,
                                 _Float16* __restrict__ Ch, int ldh,
                                 int N, int K)
{
    const int lane   = threadIdx.x & 31;
    const int wave   = threadIdx.x >> 5;
    const int ntiles = N >> 4;
    const int npb    = 8 * NTW;                 // n-tiles per block
    const int bpr    = ntiles / npb;            // blocks per m-row
    const int m0     = (blockIdx.x / bpr) << 4;
    const int nt0    = (blockIdx.x % bpr) * npb + wave * NTW;

    const int hl   = lane & 15;
    const int sel  = (lane < 16) ? 0 : 8;       // A: K sub-select per half-wave
    const int g    = (lane < 16) ? 0 : 16;      // unused; B loads also use sel
    const int mrow = m0 + hl;
    const int mbase = m0 + ((lane < 16) ? 0 : 8);

    float8 c[NTW];
    if (ACC) {
#pragma unroll
        for (int j = 0; j < NTW; ++j) {
            const int ncol = ((nt0 + j) << 4) + hl;
#pragma unroll
            for (int r = 0; r < 8; ++r)
                c[j][r] = Cf[(size_t)(mbase + r) * ldc + ncol];
        }
    } else {
#pragma unroll
        for (int j = 0; j < NTW; ++j)
#pragma unroll
            for (int r = 0; r < 8; ++r) c[j][r] = 0.0f;
    }

    for (int k0 = 0; k0 < K; k0 += 32) {
        // ---- A fragment: 16x32 f16 (M x K), documented lane layout ----
        half16 a;
        if (AF32) {
            const float* A  = (const float*)Aptr;
            const float* p0 = A + (size_t)mrow * lda + k0 + sel;
            const float* p1 = A + (size_t)mrow * lda + k0 + 16 + sel;
            __builtin_prefetch(p0 + 32, 0, 0);   // next K chunk
            fp32x4 a00 = *(const fp32x4*)(p0);
            fp32x4 a01 = *(const fp32x4*)(p0 + 4);
            fp32x4 a10 = *(const fp32x4*)(p1);
            fp32x4 a11 = *(const fp32x4*)(p1 + 4);
#pragma unroll
            for (int i = 0; i < 4; ++i) {
                a[i]      = (_Float16)a00[i];
                a[4 + i]  = (_Float16)a01[i];
                a[8 + i]  = (_Float16)a10[i];
                a[12 + i] = (_Float16)a11[i];
            }
        } else {
            const _Float16* A = (const _Float16*)Aptr;
            const _Float16* p = A + (size_t)mrow * lda + k0 + sel;
            __builtin_prefetch(p + 32, 0, 0);    // next K chunk
            half8 lo = *(const half8*)(p);
            half8 hi = *(const half8*)(p + 16);
#pragma unroll
            for (int i = 0; i < 8; ++i) { a[i] = lo[i]; a[8 + i] = hi[i]; }
        }

        // ---- B fragments: 32x16 f16 (K x N); W is (N,K) K-contiguous, so
        // each lane needs halfs [sel,sel+8) and [16+sel,24+sel) of row ncol -> two b128 loads.
#pragma unroll
        for (int j = 0; j < NTW; ++j) {
            const int ncol = ((nt0 + j) << 4) + hl;
            const _Float16* Bp = Bw + (size_t)ncol * ldb + k0 + sel;
            half8 b0 = *(const half8*)(Bp);
            half8 b1 = *(const half8*)(Bp + 16);
            half16 b;
#pragma unroll
            for (int i = 0; i < 8; ++i) { b[i] = b0[i]; b[8 + i] = b1[i]; }
            c[j] = __builtin_amdgcn_wmma_f32_16x16x32_f16(
                       false, a, false, b, (short)0, c[j], false, false);
        }
    }

#pragma unroll
    for (int j = 0; j < NTW; ++j) {
        const int ncol = ((nt0 + j) << 4) + hl;
        const float bv = bias ? bias[ncol] : 0.0f;
#pragma unroll
        for (int r = 0; r < 8; ++r) {
            float v = c[j][r] + bv;
            const size_t m = (size_t)(mbase + r);
            if (Cf) Cf[m * ldc + ncol] = v;
            if (Ch) Ch[m * ldh + ncol] = (_Float16)v;
        }
    }
}

template <int AF32, int NTW, int ACC>
static inline void launch_gemm(const void* A, int lda,
                               const _Float16* Bw, int ldb, const float* bias,
                               float* Cf, int ldc, _Float16* Ch, int ldh,
                               int M, int N, int K, hipStream_t stream)
{
    const int bpr  = (N / 16) / (8 * NTW);
    const int grid = (M / 16) * bpr;
    wmma_gemm_kernel<AF32, NTW, ACC><<<grid, 256, 0, stream>>>(
        A, lda, Bw, ldb, bias, Cf, ldc, Ch, ldh, N, K);
}

// ---------------------------------------------------------------------------
// Attention: per (b,h) block. scores = q.K^T * scale, softmax, o = a.V
// K/V stored f16 as (B*S, H) row-major -> K[(b*S+s)*H + h*64 + d]
// ---------------------------------------------------------------------------
__global__ void attn_kernel(const float* __restrict__ q,
                            const _Float16* __restrict__ K,
                            const _Float16* __restrict__ V,
                            _Float16* __restrict__ o, float scale)
{
    const int bh = blockIdx.x;
    const int b = bh >> 3;
    const int h = bh & 7;
    const int tid = threadIdx.x;

    __shared__ float qs[HDd];
    __shared__ float sc[Ssz];
    __shared__ float red[256];
    __shared__ float part[4][HDd];

    if (tid < HDd) qs[tid] = q[(size_t)b * Hsz + h * HDd + tid];
    __syncthreads();

    // scores
    for (int s = tid; s < Ssz; s += 256) {
        const _Float16* kp = K + ((size_t)(b * Ssz + s) * Hsz + h * HDd);
        float acc = 0.0f;
#pragma unroll
        for (int i = 0; i < 8; ++i) {
            half8 kv = *(const half8*)(kp + i * 8);
#pragma unroll
            for (int j = 0; j < 8; ++j) acc += qs[i * 8 + j] * (float)kv[j];
        }
        sc[s] = acc * scale;
    }
    __syncthreads();

    // softmax: max
    float lm = -1e30f;
    for (int s = tid; s < Ssz; s += 256) lm = fmaxf(lm, sc[s]);
    red[tid] = lm; __syncthreads();
    for (int st = 128; st > 0; st >>= 1) {
        if (tid < st) red[tid] = fmaxf(red[tid], red[tid + st]);
        __syncthreads();
    }
    const float mx = red[0];
    __syncthreads();

    // softmax: exp + sum
    float ls = 0.0f;
    for (int s = tid; s < Ssz; s += 256) {
        float e = expf(sc[s] - mx);
        sc[s] = e;
        ls += e;
    }
    red[tid] = ls; __syncthreads();
    for (int st = 128; st > 0; st >>= 1) {
        if (tid < st) red[tid] += red[tid + st];
        __syncthreads();
    }
    const float inv = 1.0f / red[0];
    __syncthreads();

    // weighted sum over V: thread (g,d), g in 0..3 covers 128 s values
    const int d = tid & 63;
    const int gg = tid >> 6;
    float accv = 0.0f;
    const int s0 = gg * 128;
    for (int s = s0; s < s0 + 128; ++s)
        accv += sc[s] * (float)V[(size_t)(b * Ssz + s) * Hsz + h * HDd + d];
    part[gg][d] = accv;
    __syncthreads();

    if (tid < HDd) {
        float sum = (part[0][tid] + part[1][tid] + part[2][tid] + part[3][tid]) * inv;
        o[(size_t)b * Hsz + h * HDd + tid] = (_Float16)sum;
    }
}

// ---------------------------------------------------------------------------
// LSTM pointwise: gates (B,4H) f32 -> h (f16 [+f32]), c in-place
// ---------------------------------------------------------------------------
__global__ void lstm_pw_kernel(const float* __restrict__ gates,
                               float* __restrict__ c,
                               _Float16* __restrict__ h16,
                               float* __restrict__ h32, int BH)
{
    const int i = blockIdx.x * blockDim.x + threadIdx.x;
    if (i >= BH) return;
    const int b = i >> 9;
    const int j = i & 511;
    const float* g = gates + (size_t)b * (4 * Hsz);
    const float ig = g[j], fg = g[Hsz + j], ggv = g[2 * Hsz + j], og = g[3 * Hsz + j];
    const float si = 1.0f / (1.0f + expf(-ig));
    const float sf = 1.0f / (1.0f + expf(-fg));
    const float so = 1.0f / (1.0f + expf(-og));
    const float cn = sf * c[i] + si * tanhf(ggv);
    const float hn = so * tanhf(cn);
    c[i] = cn;
    h16[i] = (_Float16)hn;
    if (h32) h32[i] = hn;
}

// ---------------------------------------------------------------------------
// Regression head: out[b, t] = dot(h1[b,:], Wreg) + breg. One wave per b.
// ---------------------------------------------------------------------------
__global__ void reg_kernel(const float* __restrict__ h1,
                           const float* __restrict__ Wreg,
                           const float* __restrict__ breg,
                           float* __restrict__ out, int t)
{
    const int lane = threadIdx.x & 31;
    const int wave = threadIdx.x >> 5;
    const int b = blockIdx.x * 8 + wave;
    float acc = 0.0f;
    for (int i = lane; i < Hsz; i += 32)
        acc += h1[(size_t)b * Hsz + i] * Wreg[i];
    for (int off = 16; off > 0; off >>= 1)
        acc += __shfl_down(acc, off, 32);
    if (lane == 0) out[(size_t)b * Tst + t] = acc + breg[0];
}

// ---------------------------------------------------------------------------
// Init: h/c state buffers, f16 h copies, y_t half of concat buffer x
// ---------------------------------------------------------------------------
__global__ void init_kernel(const float* __restrict__ enc,
                            const float* __restrict__ h0in,
                            const float* __restrict__ c0in,
                            _Float16* __restrict__ h0_16,
                            _Float16* __restrict__ h1_16,
                            float* __restrict__ h1_32,
                            float* __restrict__ c0f,
                            float* __restrict__ c1f,
                            _Float16* __restrict__ x16)
{
    const int i = blockIdx.x * blockDim.x + threadIdx.x;
    const int BH = Bsz * Hsz;
    if (i >= BH) return;
    const int b = i >> 9;
    const int j = i & 511;
    const float a0 = h0in[i];
    const float a1 = h0in[BH + i];
    h0_16[i] = (_Float16)a0;
    h1_16[i] = (_Float16)a1;
    h1_32[i] = a1;
    c0f[i] = c0in[i];
    c1f[i] = c0in[BH + i];
    x16[(size_t)b * (2 * Hsz) + j] = (_Float16)enc[((size_t)b * Ssz + (Ssz - 1)) * Hsz + j];
}

__global__ void f32_to_f16_kernel(const float* __restrict__ src,
                                  _Float16* __restrict__ dst, int n)
{
    const int i = blockIdx.x * blockDim.x + threadIdx.x;
    if (i < n) dst[i] = (_Float16)src[i];
}

// ---------------------------------------------------------------------------
extern "C" void kernel_launch(void* const* d_in, const int* in_sizes, int n_in,
                              void* d_out, int out_size, void* d_ws, size_t ws_size,
                              hipStream_t stream)
{
    const float* enc   = (const float*)d_in[0];
    const float* h0in  = (const float*)d_in[1];
    const float* c0in  = (const float*)d_in[2];
    const float* Wq    = (const float*)d_in[3];
    const float* bq    = (const float*)d_in[4];
    const float* Wk    = (const float*)d_in[5];
    const float* bk    = (const float*)d_in[6];
    const float* Wv    = (const float*)d_in[7];
    const float* bv    = (const float*)d_in[8];
    const float* Wo    = (const float*)d_in[9];
    const float* bo    = (const float*)d_in[10];
    const float* Wih0  = (const float*)d_in[11];
    const float* Whh0  = (const float*)d_in[12];
    const float* bih0  = (const float*)d_in[13];
    const float* bhh0  = (const float*)d_in[14];
    const float* Wih1  = (const float*)d_in[15];
    const float* Whh1  = (const float*)d_in[16];
    const float* bih1  = (const float*)d_in[17];
    const float* bhh1  = (const float*)d_in[18];
    const float* Wreg  = (const float*)d_in[19];
    const float* breg  = (const float*)d_in[20];
    float* out = (float*)d_out;

    // ---- workspace carve-up (256B aligned) ----
    char* ws = (char*)d_ws;
    size_t off = 0;
    auto carve = [&](size_t bytes) {
        char* p = ws + off;
        off += (bytes + 255) & ~(size_t)255;
        return p;
    };
    const size_t BS = (size_t)Bsz * Ssz;          // 131072
    const size_t BH = (size_t)Bsz * Hsz;          // 131072
    _Float16* Kf16   = (_Float16*)carve(BS * Hsz * 2);   // 128 MB
    _Float16* Vf16   = (_Float16*)carve(BS * Hsz * 2);   // 128 MB
    _Float16* Wq16   = (_Float16*)carve((size_t)Hsz * Hsz * 2);
    _Float16* Wk16   = (_Float16*)carve((size_t)Hsz * Hsz * 2);
    _Float16* Wv16   = (_Float16*)carve((size_t)Hsz * Hsz * 2);
    _Float16* Wo16   = (_Float16*)carve((size_t)Hsz * Hsz * 2);
    _Float16* Wih016 = (_Float16*)carve((size_t)4 * Hsz * 2 * Hsz * 2);
    _Float16* Whh016 = (_Float16*)carve((size_t)4 * Hsz * Hsz * 2);
    _Float16* Wih116 = (_Float16*)carve((size_t)4 * Hsz * Hsz * 2);
    _Float16* Whh116 = (_Float16*)carve((size_t)4 * Hsz * Hsz * 2);
    _Float16* h0_16  = (_Float16*)carve(BH * 2);
    _Float16* h1_16  = (_Float16*)carve(BH * 2);
    float*    h1_32  = (float*)carve(BH * 4);
    float*    c0f    = (float*)carve(BH * 4);
    float*    c1f    = (float*)carve(BH * 4);
    float*    q32    = (float*)carve(BH * 4);
    _Float16* o16    = (_Float16*)carve(BH * 2);
    _Float16* x16    = (_Float16*)carve((size_t)Bsz * 2 * Hsz * 2);
    float*    gates  = (float*)carve((size_t)Bsz * 4 * Hsz * 4);
    (void)ws_size; (void)in_sizes; (void)n_in; (void)out_size;

    auto cvt = [&](const float* s, _Float16* d, int n) {
        f32_to_f16_kernel<<<(n + 255) / 256, 256, 0, stream>>>(s, d, n);
    };

    // ---- one-time: convert weights, init state ----
    cvt(Wq, Wq16, Hsz * Hsz);
    cvt(Wk, Wk16, Hsz * Hsz);
    cvt(Wv, Wv16, Hsz * Hsz);
    cvt(Wo, Wo16, Hsz * Hsz);
    cvt(Wih0, Wih016, 4 * Hsz * 2 * Hsz);
    cvt(Whh0, Whh016, 4 * Hsz * Hsz);
    cvt(Wih1, Wih116, 4 * Hsz * Hsz);
    cvt(Whh1, Whh116, 4 * Hsz * Hsz);
    init_kernel<<<(int)(BH / 256), 256, 0, stream>>>(
        enc, h0in, c0in, h0_16, h1_16, h1_32, c0f, c1f, x16);

    // ---- K/V projections: f32 A read directly, f16 tiles stored ----
    // AF32=1, 4 N-tiles per wave (register A reuse), no accumulate.
    launch_gemm<1, 4, 0>(enc, Hsz, Wk16, Hsz, bk, nullptr, 0, Kf16, Hsz,
                         (int)BS, Hsz, Hsz, stream);
    launch_gemm<1, 4, 0>(enc, Hsz, Wv16, Hsz, bv, nullptr, 0, Vf16, Hsz,
                         (int)BS, Hsz, Hsz, stream);

    const float scale = 1.0f / 8.0f; // 1/sqrt(64)

    for (int t = 0; t < Tst; ++t) {
        // q = h1 @ Wq^T + bq  (f32 out)
        launch_gemm<0, 1, 0>(h1_16, Hsz, Wq16, Hsz, bq, q32, Hsz, nullptr, 0,
                             Bsz, Hsz, Hsz, stream);
        // attention -> o16
        attn_kernel<<<Bsz * NHh, 256, 0, stream>>>(q32, Kf16, Vf16, o16, scale);
        // attn_out = o @ Wo^T + bo, written f16 directly into x[:, 512:1024]
        launch_gemm<0, 1, 0>(o16, Hsz, Wo16, Hsz, bo, nullptr, 0, x16 + Hsz, 2 * Hsz,
                             Bsz, Hsz, Hsz, stream);
        // gates0 = x @ Wih0^T + bih0 + h0 @ Whh0^T + bhh0
        launch_gemm<0, 1, 0>(x16, 2 * Hsz, Wih016, 2 * Hsz, bih0, gates, 4 * Hsz,
                             nullptr, 0, Bsz, 4 * Hsz, 2 * Hsz, stream);
        launch_gemm<0, 1, 1>(h0_16, Hsz, Whh016, Hsz, bhh0, gates, 4 * Hsz,
                             nullptr, 0, Bsz, 4 * Hsz, Hsz, stream);
        lstm_pw_kernel<<<(int)(BH / 256), 256, 0, stream>>>(
            gates, c0f, h0_16, nullptr, (int)BH);
        // gates1 = h0n @ Wih1^T + bih1 + h1 @ Whh1^T + bhh1
        launch_gemm<0, 1, 0>(h0_16, Hsz, Wih116, Hsz, bih1, gates, 4 * Hsz,
                             nullptr, 0, Bsz, 4 * Hsz, Hsz, stream);
        launch_gemm<0, 1, 1>(h1_16, Hsz, Whh116, Hsz, bhh1, gates, 4 * Hsz,
                             nullptr, 0, Bsz, 4 * Hsz, Hsz, stream);
        lstm_pw_kernel<<<(int)(BH / 256), 256, 0, stream>>>(
            gates, c1f, h1_16, h1_32, (int)BH);
        // y[:, t] = h1 @ Wreg^T + breg
        reg_kernel<<<Bsz / 8, 256, 0, stream>>>(h1_32, Wreg, breg, out, t);
    }
}
